// DSVDD_33397665693701
// MI455X (gfx1250) — matrix operand order repaired and checked by reading, not compiled, not run on hardware
//
#include <hip/hip_runtime.h>
#include <math.h>

// ---------------------------------------------------------------------------
// DSVDD-style scoring on MI455X (gfx1250, wave32, WMMA).
//   phi  = einsum('bchw,dc->bdhw', p, W) + b          (GEMM1, 80.6 GFLOP)
//   f_c  = 2 * phi_p @ C                              (GEMM2, 141 GFLOP)
//   dist = sqrt(||phi||^2 + ||c||^2 - f_c); fused top-3 softmin score.
// Precision: bf16 hi/lo split ("bf16x3": hi*hi + hi*lo + lo*hi, f32 acc)
// on V_WMMA_F32_16X16X32_BF16 — near-fp32 accuracy at bf16 matrix rates.
// phi is kept in pre-split bf16 hi/lo form end-to-end so GEMM2 can stage its
// A-tiles into LDS with GLOBAL_LOAD_ASYNC_TO_LDS (ASYNCcnt DMA) when the
// toolchain exposes the builtin (guarded; manual v8us copy fallback).
// ---------------------------------------------------------------------------

#define DIMK   1792
#define HW     3136
#define MROWS  12544     // 4 * 3136
#define NPROT  3136
#define PITCH  1800      // LDS row pitch in ushorts (bank-conflict padding)

typedef __attribute__((ext_vector_type(16))) __bf16        v16bf;
typedef __attribute__((ext_vector_type(16))) unsigned short v16us;
typedef __attribute__((ext_vector_type(8)))  unsigned short v8us;
typedef __attribute__((ext_vector_type(8)))  float          v8f;
typedef __attribute__((ext_vector_type(4)))  int            v4i;

typedef __attribute__((address_space(1))) v4i* as1_v4i_ptr;  // global int4*
typedef __attribute__((address_space(3))) v4i* as3_v4i_ptr;  // LDS int4* (32-bit)

#if defined(__has_builtin)
#if __has_builtin(__builtin_amdgcn_global_load_async_to_lds_b128)
#define HAVE_ASYNC_LDS 1
#endif
#endif

__device__ __forceinline__ void splitf(float x, unsigned short& h, unsigned short& l) {
    unsigned u  = __float_as_uint(x);
    unsigned hr = (u + 0x8000u) & 0xFFFF0000u;   // round-to-nearest bf16 (hi)
    float    hf = __uint_as_float(hr);
    float    lf = x - hf;                        // residual
    unsigned ul = __float_as_uint(lf);
    h = (unsigned short)(hr >> 16);
    l = (unsigned short)((ul + 0x8000u) >> 16);
}

__device__ __forceinline__ float bf2pair_to_f32(unsigned short h, unsigned short l) {
    return __uint_as_float((unsigned)h << 16) + __uint_as_float((unsigned)l << 16);
}

__device__ __forceinline__ v8f wmma_bf16(v16us a, v16us b, v8f c) {
    return __builtin_amdgcn_wmma_f32_16x16x32_bf16(
        false, __builtin_bit_cast(v16bf, a),
        false, __builtin_bit_cast(v16bf, b),
        (short)0, c, false, false);
}

__device__ __forceinline__ v8us ld8(const unsigned short* p) {
    return *(const v8us*)p;
}

__device__ __forceinline__ v16us cat8(v8us lo, v8us hi) {
    v16us r;
#pragma unroll
    for (int i = 0; i < 8; ++i) { r[i] = lo[i]; r[i + 8] = hi[i]; }
    return r;
}

#ifdef HAVE_ASYNC_LDS
// 16B global -> LDS async copy (VGPR-bypassing, tracked by ASYNCcnt).
// AS3 pointers are 32-bit on amdgcn; low 32 bits of a generic LDS address are
// the LDS offset.
__device__ __forceinline__ void async_cp16(const unsigned short* g, unsigned short* l) {
    __builtin_amdgcn_global_load_async_to_lds_b128(
        (as1_v4i_ptr)(size_t)g,
        (as3_v4i_ptr)(unsigned int)(size_t)l,
        0, 0);
}
__device__ __forceinline__ void wait_async0() {
#if __has_builtin(__builtin_amdgcn_s_wait_asynccnt)
    __builtin_amdgcn_s_wait_asynccnt(0);
#else
    asm volatile("s_wait_asynccnt 0" ::: "memory");
#endif
}
#endif

// maintain ascending triple (t0 <= t1 <= t2)
__device__ __forceinline__ void ins3(float& t0, float& t1, float& t2, float v) {
    if (v < t2) {
        if (v < t1) {
            t2 = t1;
            if (v < t0) { t1 = t0; t0 = v; } else { t1 = v; }
        } else {
            t2 = v;
        }
    }
}

// --------------------------- prep kernels ----------------------------------

__global__ __launch_bounds__(256) void split_kernel(
    const float* __restrict__ src, unsigned short* __restrict__ hi,
    unsigned short* __restrict__ lo, int n)
{
    int stride = gridDim.x * blockDim.x;
    for (int i = blockIdx.x * blockDim.x + threadIdx.x; i < n; i += stride) {
        unsigned short h, l;
        splitf(src[i], h, l);
        hi[i] = h; lo[i] = l;
    }
}

// C is [DIM, P]; emit transposed+split Ct[p][d] so GEMM2 B-fragments are
// contiguous 32B vector loads. Writes coalesced; strided reads hit L2 (C=22MB).
__global__ __launch_bounds__(256) void splitC_t_kernel(
    const float* __restrict__ C, unsigned short* __restrict__ hiT,
    unsigned short* __restrict__ loT)
{
    int stride = gridDim.x * blockDim.x;
    for (int i = blockIdx.x * blockDim.x + threadIdx.x; i < DIMK * NPROT; i += stride) {
        int pcol = i / DIMK;
        int d    = i - pcol * DIMK;
        unsigned short h, l;
        splitf(C[(size_t)d * NPROT + pcol], h, l);
        hiT[i] = h; loT[i] = l;
    }
}

__global__ __launch_bounds__(256) void centers_kernel(
    const float* __restrict__ C, float* __restrict__ cen)
{
    int p = blockIdx.x * blockDim.x + threadIdx.x;
    if (p >= NPROT) return;
    float s = 0.f;
    for (int d = 0; d < DIMK; ++d) {
        float v = C[(size_t)d * NPROT + p];
        s += v * v;
    }
    cen[p] = s;
}

// ||phi_m||^2 from the pre-split hi/lo representation (exact to bf16x2).
__global__ __launch_bounds__(256) void features_kernel(
    const unsigned short* __restrict__ phiH, const unsigned short* __restrict__ phiL,
    float* __restrict__ feats)
{
    __shared__ float red[256];
    const int m = blockIdx.x;
    const unsigned short* rh = phiH + (size_t)m * DIMK;
    const unsigned short* rl = phiL + (size_t)m * DIMK;
    float s = 0.f;
    for (int c = threadIdx.x; c < DIMK; c += 256) {
        float v = bf2pair_to_f32(rh[c], rl[c]);
        s += v * v;
    }
    red[threadIdx.x] = s;
    __syncthreads();
    for (int off = 128; off > 0; off >>= 1) {
        if (threadIdx.x < off) red[threadIdx.x] += red[threadIdx.x + off];
        __syncthreads();
    }
    if (threadIdx.x == 0) feats[m] = red[0];
}

// --------------------------- GEMM1: phi ------------------------------------
// grid (392, 14), block 256 (8 waves). Block = 32 m-rows; wave = one 16-col
// n-tile x two stacked 16-row m-tiles (B-operand reuse x2). A (p slice) staged
// in LDS as split bf16 hi/lo once per block. Output written pre-split.

__global__ __launch_bounds__(256) void gemm1_phi(
    const float* __restrict__ p, const unsigned short* __restrict__ Whi,
    const unsigned short* __restrict__ Wlo, const float* __restrict__ bias,
    unsigned short* __restrict__ phiH, unsigned short* __restrict__ phiL)
{
    extern __shared__ unsigned char smem_raw[];
    unsigned short* sAhi = (unsigned short*)smem_raw;     // [32][PITCH]
    unsigned short* sAlo = sAhi + 32 * PITCH;

    const int m0  = blockIdx.x * 32;                      // 3136 % 32 == 0
    const int b   = m0 / HW;
    const int hw0 = m0 - b * HW;
    const int tid = threadIdx.x;

    for (int idx = tid; idx < 32 * DIMK; idx += 256) {
        int c  = idx >> 5;
        int mr = idx & 31;
        float x = p[((size_t)b * DIMK + c) * HW + hw0 + mr];
        unsigned short h, l;
        splitf(x, h, l);
        sAhi[mr * PITCH + c] = h;
        sAlo[mr * PITCH + c] = l;
    }
    __syncthreads();

    const int wave  = tid >> 5;
    const int lane  = tid & 31;
    const int lhalf = lane & 15;
    const int hi16  = lane >> 4;

    const int n0  = (blockIdx.y * 8 + wave) * 16;
    const int d_g = n0 + lhalf;

    const unsigned short* wh  = Whi + (size_t)d_g * DIMK + hi16 * 16;
    const unsigned short* wl  = Wlo + (size_t)d_g * DIMK + hi16 * 16;
    const unsigned short* ah0 = sAhi + lhalf * PITCH + hi16 * 16;
    const unsigned short* al0 = sAlo + lhalf * PITCH + hi16 * 16;
    const unsigned short* ah1 = ah0 + 16 * PITCH;
    const unsigned short* al1 = al0 + 16 * PITCH;

    v8f acc0 = {};
    v8f acc1 = {};
    for (int kk = 0; kk < DIMK; kk += 32) {
        v16us bh  = *(const v16us*)(wh + kk);
        v16us bl  = *(const v16us*)(wl + kk);
        v16us a0h = cat8(ld8(ah0 + kk), ld8(ah0 + kk + 8));
        v16us a0l = cat8(ld8(al0 + kk), ld8(al0 + kk + 8));
        v16us a1h = cat8(ld8(ah1 + kk), ld8(ah1 + kk + 8));
        v16us a1l = cat8(ld8(al1 + kk), ld8(al1 + kk + 8));
        acc0 = wmma_bf16(a0h, bh, acc0);
        acc0 = wmma_bf16(a0h, bl, acc0);
        acc0 = wmma_bf16(a0l, bh, acc0);
        acc1 = wmma_bf16(a1h, bh, acc1);
        acc1 = wmma_bf16(a1h, bl, acc1);
        acc1 = wmma_bf16(a1l, bh, acc1);
    }

    const float bv = bias[d_g];
#pragma unroll
    for (int r = 0; r < 8; ++r) {
        int mr = r + hi16 * 8;
        unsigned short h, l;
        splitf(acc0[r] + bv, h, l);
        size_t i0 = (size_t)(m0 + mr) * DIMK + d_g;
        phiH[i0] = h; phiL[i0] = l;
        splitf(acc1[r] + bv, h, l);
        size_t i1 = (size_t)(m0 + 16 + mr) * DIMK + d_g;
        phiH[i1] = h; phiL[i1] = l;
    }
}

// ------------------ GEMM2 + fused top-3 softmin score ----------------------
// grid 392, block 256 (8 waves). Block = 32 m-rows; waves stride the 196
// prototype n-tiles. A-tiles staged into LDS via async global->LDS DMA
// (pre-split bf16, pure byte copy). Running per-lane top-3 -> shfl butterfly
// within 16-lane halves -> LDS merge across waves -> softmin score.

__global__ __launch_bounds__(256) void gemm2_score(
    const unsigned short* __restrict__ phiH, const unsigned short* __restrict__ phiL,
    const unsigned short* __restrict__ ChiT, const unsigned short* __restrict__ CloT,
    const float* __restrict__ feats, const float* __restrict__ cen,
    float* __restrict__ out)
{
    extern __shared__ unsigned char smem_raw[];
    unsigned short* sAhi = (unsigned short*)smem_raw;     // [32][PITCH]
    unsigned short* sAlo = sAhi + 32 * PITCH;
    float* sTop = (float*)(sAlo + 32 * PITCH);            // [8 waves][32 rows][3]

    const int m0  = blockIdx.x * 32;
    const int tid = threadIdx.x;

    // stage A (32 rows x 1792 bf16 hi/lo) into padded LDS, 16B chunks
    const int CHUNKS = DIMK / 8;                          // 224 per row
    for (int idx = tid; idx < 32 * CHUNKS; idx += 256) {
        int mr = idx / CHUNKS;
        int cc = idx - mr * CHUNKS;
        const unsigned short* gh = phiH + (size_t)(m0 + mr) * DIMK + cc * 8;
        const unsigned short* gl = phiL + (size_t)(m0 + mr) * DIMK + cc * 8;
        unsigned short* dh = sAhi + mr * PITCH + cc * 8;
        unsigned short* dl = sAlo + mr * PITCH + cc * 8;
#ifdef HAVE_ASYNC_LDS
        async_cp16(gh, dh);
        async_cp16(gl, dl);
#else
        *(v8us*)dh = *(const v8us*)gh;
        *(v8us*)dl = *(const v8us*)gl;
#endif
    }
#ifdef HAVE_ASYNC_LDS
    wait_async0();
#endif
    __syncthreads();

    const int wave  = tid >> 5;
    const int lane  = tid & 31;
    const int lhalf = lane & 15;
    const int hi16  = lane >> 4;

    const unsigned short* ah0 = sAhi + lhalf * PITCH + hi16 * 16;
    const unsigned short* al0 = sAlo + lhalf * PITCH + hi16 * 16;
    const unsigned short* ah1 = ah0 + 16 * PITCH;
    const unsigned short* al1 = al0 + 16 * PITCH;

    float feat0[8], feat1[8];
#pragma unroll
    for (int r = 0; r < 8; ++r) {
        feat0[r] = feats[m0 + r + hi16 * 8];
        feat1[r] = feats[m0 + 16 + r + hi16 * 8];
    }

    const float INF = 3.0e38f;
    float t0a[8], t1a[8], t2a[8], t0b[8], t1b[8], t2b[8];
#pragma unroll
    for (int r = 0; r < 8; ++r) {
        t0a[r] = t1a[r] = t2a[r] = INF;
        t0b[r] = t1b[r] = t2b[r] = INF;
    }

    for (int nt = wave; nt < NPROT / 16; nt += 8) {
        const int p_g = nt * 16 + lhalf;
        const float c2 = cen[p_g];
        const unsigned short* ch = ChiT + (size_t)p_g * DIMK + hi16 * 16;
        const unsigned short* cl = CloT + (size_t)p_g * DIMK + hi16 * 16;

        v8f acc0 = {};
        v8f acc1 = {};
        for (int kk = 0; kk < DIMK; kk += 32) {
            v16us bh  = *(const v16us*)(ch + kk);
            v16us bl  = *(const v16us*)(cl + kk);
            v16us a0h = cat8(ld8(ah0 + kk), ld8(ah0 + kk + 8));
            v16us a0l = cat8(ld8(al0 + kk), ld8(al0 + kk + 8));
            v16us a1h = cat8(ld8(ah1 + kk), ld8(ah1 + kk + 8));
            v16us a1l = cat8(ld8(al1 + kk), ld8(al1 + kk + 8));
            acc0 = wmma_bf16(a0h, bh, acc0);
            acc0 = wmma_bf16(a0h, bl, acc0);
            acc0 = wmma_bf16(a0l, bh, acc0);
            acc1 = wmma_bf16(a1h, bh, acc1);
            acc1 = wmma_bf16(a1h, bl, acc1);
            acc1 = wmma_bf16(a1l, bh, acc1);
        }
#pragma unroll
        for (int r = 0; r < 8; ++r) {
            float d2 = feat0[r] + c2 - 2.0f * acc0[r];
            ins3(t0a[r], t1a[r], t2a[r], sqrtf(fmaxf(d2, 0.0f)));
            d2 = feat1[r] + c2 - 2.0f * acc1[r];
            ins3(t0b[r], t1b[r], t2b[r], sqrtf(fmaxf(d2, 0.0f)));
        }
    }

    // butterfly top-3 merge across the 16 lanes of each half-wave
#pragma unroll
    for (int off = 1; off < 16; off <<= 1) {
#pragma unroll
        for (int r = 0; r < 8; ++r) {
            float o0 = __shfl_xor(t0a[r], off, 16);
            float o1 = __shfl_xor(t1a[r], off, 16);
            float o2 = __shfl_xor(t2a[r], off, 16);
            ins3(t0a[r], t1a[r], t2a[r], o0);
            ins3(t0a[r], t1a[r], t2a[r], o1);
            ins3(t0a[r], t1a[r], t2a[r], o2);
            o0 = __shfl_xor(t0b[r], off, 16);
            o1 = __shfl_xor(t1b[r], off, 16);
            o2 = __shfl_xor(t2b[r], off, 16);
            ins3(t0b[r], t1b[r], t2b[r], o0);
            ins3(t0b[r], t1b[r], t2b[r], o1);
            ins3(t0b[r], t1b[r], t2b[r], o2);
        }
    }

    if (lhalf == 0) {
#pragma unroll
        for (int r = 0; r < 8; ++r) {
            int rowA = r + hi16 * 8;
            int rowB = 16 + rowA;
            float* qa = &sTop[(wave * 32 + rowA) * 3];
            qa[0] = t0a[r]; qa[1] = t1a[r]; qa[2] = t2a[r];
            float* qb = &sTop[(wave * 32 + rowB) * 3];
            qb[0] = t0b[r]; qb[1] = t1b[r]; qb[2] = t2b[r];
        }
    }
    __syncthreads();

    if (tid < 32) {
        float b0 = INF, b1 = INF, b2 = INF;
        for (int w = 0; w < 8; ++w) {
            const float* q = &sTop[(w * 32 + tid) * 3];
            ins3(b0, b1, b2, q[0]);
            ins3(b0, b1, b2, q[1]);
            ins3(b0, b1, b2, q[2]);
        }
        // softmin weight of nearest prototype times its distance
        float w0 = 1.0f / (1.0f + expf(b0 - b1) + expf(b0 - b2));
        out[m0 + tid] = w0 * b0;
    }
}

// --------------------------- launcher --------------------------------------

extern "C" void kernel_launch(void* const* d_in, const int* in_sizes, int n_in,
                              void* d_out, int out_size, void* d_ws, size_t ws_size,
                              hipStream_t stream) {
    (void)in_sizes; (void)n_in; (void)out_size; (void)ws_size;
    const float* p    = (const float*)d_in[0];   // [4,1792,56,56]
    const float* W    = (const float*)d_in[1];   // [1792,1792]
    const float* bias = (const float*)d_in[2];   // [1792]
    const float* C    = (const float*)d_in[3];   // [1792,3136]
    float* out        = (float*)d_out;           // [4,1,56,56] -> 12544

    char*  base = (char*)d_ws;
    size_t off  = 0;
    auto take = [&](size_t bytes) -> char* {
        char* r = base + off;
        off += (bytes + 255) & ~(size_t)255;
        return r;
    };
    unsigned short* phiH = (unsigned short*)take((size_t)MROWS * DIMK * 2); // 45 MB
    unsigned short* phiL = (unsigned short*)take((size_t)MROWS * DIMK * 2); // 45 MB
    float* feats = (float*)take((size_t)MROWS * sizeof(float));
    float* cen   = (float*)take((size_t)NPROT * sizeof(float));
    unsigned short* Whi  = (unsigned short*)take((size_t)DIMK * DIMK * 2);  // 6.4 MB
    unsigned short* Wlo  = (unsigned short*)take((size_t)DIMK * DIMK * 2);
    unsigned short* ChiT = (unsigned short*)take((size_t)DIMK * NPROT * 2); // 11.2 MB
    unsigned short* CloT = (unsigned short*)take((size_t)DIMK * NPROT * 2);

    const size_t smem1 = (size_t)2 * 32 * PITCH * 2;            // 230400 B
    const size_t smem2 = smem1 + (size_t)8 * 32 * 3 * 4;        // +3072 B

    split_kernel   <<<dim3(1024), 256, 0, stream>>>(W, Whi, Wlo, DIMK * DIMK);
    splitC_t_kernel<<<dim3(2048), 256, 0, stream>>>(C, ChiT, CloT);
    centers_kernel <<<dim3((NPROT + 255) / 256), 256, 0, stream>>>(C, cen);
    gemm1_phi      <<<dim3(MROWS / 32, 14), 256, smem1, stream>>>(p, Whi, Wlo, bias, phiH, phiL);
    features_kernel<<<dim3(MROWS), 256, 0, stream>>>(phiH, phiL, feats);
    gemm2_score    <<<dim3(MROWS / 32), 256, smem2, stream>>>(phiH, phiL, ChiT, CloT, feats, cen, out);
}
